// USPAttention_38276748542529
// MI455X (gfx1250) — matrix operand order repaired and checked by reading, not compiled, not run on hardware
//
#include <hip/hip_runtime.h>

// ---------------------------------------------------------------------------
// Flash attention forward, B=1, S=4096, H=16, D=64 (fp32 in/out).
// Compute-bound (68.7 GFLOP vs ~67MB) -> v_wmma_f32_16x16x32_f16, f32 accum.
// Double-buffered LDS K/V tiles, register-pipelined staging, DPP reductions.
// ---------------------------------------------------------------------------

typedef _Float16 v16h __attribute__((ext_vector_type(16)));
typedef _Float16 h8v  __attribute__((ext_vector_type(8)));
typedef float    v8f  __attribute__((ext_vector_type(8)));

#define S_LEN   4096
#define H_NUM   16
#define D_DIM   64
#define BM      128      // query rows per block (8 waves x 16)
#define BN      64       // keys per tile
#define NTILES  (S_LEN / BN)
#define LDK     72       // padded LDS row stride in halves (144B = 36 dwords)

// exp2-domain scale: (1/sqrt(64)) * log2(e)
#define QSCALE  0.18033688011112042f

static __device__ inline v8f wmma16(v16h a, v16h b, v8f c) {
  return __builtin_amdgcn_wmma_f32_16x16x32_f16(
      false, a, false, b, (short)0, c, false, false);
}

// Load a 16-half WMMA operand as two 16B LDS chunks at p and p+delta halves.
static __device__ inline v16h ldpair(const _Float16* p, int delta) {
  h8v a = *(const h8v*)(p);
  h8v b = *(const h8v*)(p + delta);
  return __builtin_shufflevector(a, b, 0,1,2,3,4,5,6,7,8,9,10,11,12,13,14,15);
}

// Wave-local LDS producer->consumer sync (P buffer is wave-private).
static __device__ inline void wave_lds_sync() {
  __builtin_amdgcn_wave_barrier();
  asm volatile("s_wait_dscnt 0" ::: "memory");
  __builtin_amdgcn_wave_barrier();
}

// Butterfly reduction over each 16-lane DPP row (wave32 = 2 rows), pure VALU:
// quad_perm xor1 (0xB1), quad_perm xor2 (0x4E), row_half_mirror (0x141),
// row_mirror (0x140). Leaves every lane of a row holding the row result.
#define DPP_STEP(x, op, ctrl)                                                 \
  x = op(x, __int_as_float(__builtin_amdgcn_update_dpp(                       \
              __float_as_int(x), __float_as_int(x), ctrl, 0xF, 0xF, true)))

static __device__ inline float row_max16(float x) {
  DPP_STEP(x, fmaxf, 0xB1);
  DPP_STEP(x, fmaxf, 0x4E);
  DPP_STEP(x, fmaxf, 0x141);
  DPP_STEP(x, fmaxf, 0x140);
  return x;
}
static __device__ inline float faddf(float a, float b) { return a + b; }
static __device__ inline float row_sum16(float x) {
  DPP_STEP(x, faddf, 0xB1);
  DPP_STEP(x, faddf, 0x4E);
  DPP_STEP(x, faddf, 0x141);
  DPP_STEP(x, faddf, 0x140);
  return x;
}

__global__ __launch_bounds__(256)
void fa_fwd_f16wmma(const float* __restrict__ Q,
                    const float* __restrict__ K,
                    const float* __restrict__ V,
                    float* __restrict__ O) {
  __shared__ _Float16 lK [2][BN * LDK];       // [buf][key][d]   2 x 9 KB
  __shared__ _Float16 lVt[2][D_DIM * LDK];    // [buf][d][key]   2 x 9 KB
  __shared__ _Float16 lP [8 * 16 * LDK];      // per-wave P tile    18 KB

  const int tid  = threadIdx.x;
  const int lane = tid & 31;
  const int w    = tid >> 5;
  const int half = lane >> 4;
  const int lm   = lane & 15;
  const int h    = blockIdx.x & (H_NUM - 1);
  const int qt   = blockIdx.x >> 4;
  const int qbase = qt * BM + w * 16;

  // ---- Q slab (16x64) as two f16 A-operands, scale folded in (exp2 domain).
  v16h aq0, aq1;
  {
    const float* qp = Q + ((size_t)(qbase + lm) * H_NUM + h) * D_DIM;
#pragma unroll
    for (int v = 0; v < 8; ++v) {
      const int d0 = 16 * half + 2 * v;
      float2 q0 = *(const float2*)(qp + d0);
      float2 q1 = *(const float2*)(qp + d0 + 32);
      aq0[2 * v]     = (_Float16)(q0.x * QSCALE);
      aq0[2 * v + 1] = (_Float16)(q0.y * QSCALE);
      aq1[2 * v]     = (_Float16)(q1.x * QSCALE);
      aq1[2 * v + 1] = (_Float16)(q1.y * QSCALE);
    }
  }

  v8f o[4] = {};
  float mrow[8], lrow[8];
#pragma unroll
  for (int r = 0; r < 8; ++r) { mrow[r] = -1e30f; lrow[r] = 0.0f; }

  // Staging map: thread t -> key = t&63, d-quarter = t>>6 (one 64B line each).
  const int skey = tid & 63;
  const int sd   = (tid >> 6) * 16;

  // Load a tile's K/V lines into f16 registers (global -> VGPR, no LDS yet).
  auto load_tile = [&](int it2, h8v& k0, h8v& k1, h8v& v0, h8v& v1) {
    const int key0 = it2 * BN;
    const float4* kp = (const float4*)(K + ((size_t)(key0 + skey) * H_NUM + h) * D_DIM + sd);
    const float4* vp = (const float4*)(V + ((size_t)(key0 + skey) * H_NUM + h) * D_DIM + sd);
    float tf[16];
#pragma unroll
    for (int i = 0; i < 4; ++i) {
      float4 f = kp[i];
      tf[4*i] = f.x; tf[4*i+1] = f.y; tf[4*i+2] = f.z; tf[4*i+3] = f.w;
    }
#pragma unroll
    for (int i = 0; i < 8; ++i) { k0[i] = (_Float16)tf[i]; k1[i] = (_Float16)tf[8 + i]; }
#pragma unroll
    for (int i = 0; i < 4; ++i) {
      float4 f = vp[i];
      tf[4*i] = f.x; tf[4*i+1] = f.y; tf[4*i+2] = f.z; tf[4*i+3] = f.w;
    }
#pragma unroll
    for (int i = 0; i < 8; ++i) { v0[i] = (_Float16)tf[i]; v1[i] = (_Float16)tf[8 + i]; }
  };

  h8v k0, k1, v0, v1;
  load_tile(0, k0, k1, v0, v1);

  for (int it = 0; it < NTILES; ++it) {
    const int buf = it & 1;

    // ---- Commit staged registers to this iteration's LDS buffer ----
    *(h8v*)(&lK[buf][skey * LDK + sd])     = k0;
    *(h8v*)(&lK[buf][skey * LDK + sd + 8]) = k1;
#pragma unroll
    for (int i = 0; i < 8; ++i) {
      lVt[buf][(sd + i) * LDK + skey]     = v0[i];
      lVt[buf][(sd + 8 + i) * LDK + skey] = v1[i];
    }

    // ---- Issue next tile's global loads; latency hides under compute ----
    if (it + 1 < NTILES) {
      load_tile(it + 1, k0, k1, v0, v1);
      if (it + 2 < NTILES) {
        const int kn = (it + 2) * BN;
        __builtin_prefetch(K + ((size_t)(kn + skey) * H_NUM + h) * D_DIM + sd, 0, 3);
        __builtin_prefetch(V + ((size_t)(kn + skey) * H_NUM + h) * D_DIM + sd, 0, 3);
      }
    }
    __syncthreads();  // one barrier per tile (double buffering)

    // ---- S = Q K^T ----
    v8f sc[4];
#pragma unroll
    for (int j = 0; j < 4; ++j) {
      const _Float16* bp = &lK[buf][(16 * j + lm) * LDK + 16 * half];
      v16h b0 = ldpair(bp, 8);
      v16h b1 = ldpair(bp + 32, 8);
      v8f c = {};
      c = wmma16(aq0, b0, c);
      c = wmma16(aq1, b1, c);
      sc[j] = c;
    }

    // ---- Online softmax (row M = r + 8*half, replicated across 16 lanes) ----
    float al[8];
#pragma unroll
    for (int r = 0; r < 8; ++r) {
      float a = fmaxf(fmaxf(sc[0][r], sc[1][r]), fmaxf(sc[2][r], sc[3][r]));
      a = row_max16(a);
      const float mnew = fmaxf(mrow[r], a);
      al[r] = __builtin_amdgcn_exp2f(mrow[r] - mnew);
      mrow[r] = mnew;
    }

    _Float16* lPw = &lP[w * 16 * LDK];
#pragma unroll
    for (int j = 0; j < 4; ++j) {
#pragma unroll
      for (int r = 0; r < 8; ++r) {
        const float p = __builtin_amdgcn_exp2f(sc[j][r] - mrow[r]);
        sc[j][r] = p;
        lPw[(r + 8 * half) * LDK + 16 * j + lm] = (_Float16)p;
      }
    }
#pragma unroll
    for (int r = 0; r < 8; ++r) {
      float s = row_sum16(sc[0][r] + sc[1][r] + sc[2][r] + sc[3][r]);
      lrow[r] = lrow[r] * al[r] + s;
    }
#pragma unroll
    for (int j = 0; j < 4; ++j)
#pragma unroll
      for (int r = 0; r < 8; ++r) o[j][r] *= al[r];

    wave_lds_sync();  // P writes visible to this wave's transposed reads

    // ---- P as A-operands (C-layout -> A-layout via per-wave LDS tile) ----
    const _Float16* pp = &lPw[lm * LDK + 16 * half];
    v16h ap0 = ldpair(pp, 8);
    v16h ap1 = ldpair(pp + 32, 8);

    // ---- O += P V ----
#pragma unroll
    for (int j = 0; j < 4; ++j) {
      const _Float16* vb = &lVt[buf][(16 * j + lm) * LDK + 16 * half];
      v16h b0 = ldpair(vb, 8);
      v16h b1 = ldpair(vb + 32, 8);
      o[j] = wmma16(ap0, b0, o[j]);
      o[j] = wmma16(ap1, b1, o[j]);
    }
  }

  // ---- Epilogue: normalize and store ----
#pragma unroll
  for (int r = 0; r < 8; ++r) {
    const float inv = 1.0f / lrow[r];
    const int row = qbase + r + 8 * half;
    float* op = O + ((size_t)row * H_NUM + h) * D_DIM + lm;
    op[0]  = o[0][r] * inv;
    op[16] = o[1][r] * inv;
    op[32] = o[2][r] * inv;
    op[48] = o[3][r] * inv;
  }
}

extern "C" void kernel_launch(void* const* d_in, const int* in_sizes, int n_in,
                              void* d_out, int out_size, void* d_ws, size_t ws_size,
                              hipStream_t stream) {
  const float* q = (const float*)d_in[0];
  const float* k = (const float*)d_in[1];
  const float* v = (const float*)d_in[2];
  float* out = (float*)d_out;
  (void)in_sizes; (void)n_in; (void)out_size; (void)d_ws; (void)ws_size;

  const int blocks = (S_LEN / BM) * H_NUM;  // 512
  fa_fwd_f16wmma<<<blocks, 256, 0, stream>>>(q, k, v, out);
}
